// TransformerBlock_64132451663990
// MI455X (gfx1250) — matrix operand order repaired and checked by reading, not compiled, not run on hardware
//
#include <hip/hip_runtime.h>
#include <hip/hip_bf16.h>
#include <math.h>

typedef __attribute__((ext_vector_type(16))) __bf16 v16bf;
typedef __attribute__((ext_vector_type(8)))  float  v8f;
typedef __attribute__((ext_vector_type(4)))  unsigned int v4u;

#define T_D   2048
#define T_H   16
#define T_HD  128
#define T_FF  8192
#define T_B   2
#define T_S   2048
#define T_M   (T_B * T_S)

union U16BFV { v16bf v; v4u q[2]; };
union UPK    { unsigned int u; __bf16 h[2]; };

__device__ __forceinline__ v8f wmma_bf16(v16bf a, v16bf b, v8f c) {
  return __builtin_amdgcn_wmma_f32_16x16x32_bf16(false, a, false, b, (short)0, c,
                                                 false, false);
}

// CDNA5 async global->LDS copy (16B per lane), tracked by ASYNCcnt
__device__ __forceinline__ void async_b128(unsigned ldsOff, const void* g) {
  asm volatile("global_load_async_to_lds_b128 %0, %1, off"
               :: "v"(ldsOff), "v"(g) : "memory");
}
__device__ __forceinline__ void wait_async0() {
  asm volatile("s_wait_asynccnt 0x0" ::: "memory");
}

// A-fragment (16x32 bf16): lane = M row, K contiguous per half-wave
// (lanes 0-15: K=0..15, lanes 16-31: K=16..31). Two ds_load_b128.
__device__ __forceinline__ v16bf fragA_ld(const __bf16* base, int strideBf,
                                          int row0, int koff, int lane) {
  int row = row0 + (lane & 15);
  int kb  = koff + ((lane >> 4) << 4);
  const v4u* p = (const v4u*)(base + row * strideBf + kb);
  U16BFV r;
  r.q[0] = p[0];
  r.q[1] = p[1];
  return r.v;
}

// B-fragment (32x16 bf16): lane = N col, K contiguous per half-wave
// (lanes 0-15: K=0..15, lanes 16-31: K=16..31). base is [N][K]. Two b128.
__device__ __forceinline__ v16bf fragB_ld(const __bf16* base, int strideBf,
                                          int row0, int koff, int lane) {
  int row = row0 + (lane & 15);
  int kb  = koff + ((lane >> 4) << 4);
  const v4u* p = (const v4u*)(base + row * strideBf + kb);
  U16BFV r;
  r.q[0] = p[0];
  r.q[1] = p[1];
  return r.v;
}

// ---------- fp32 [K][N] -> bf16 [N][K] tiled transpose (weights) ----------
__global__ __launch_bounds__(256) void transpose_to_bf16(
    const float* __restrict__ in, __bf16* __restrict__ out, int K, int N) {
  __shared__ float tile[32][33];
  const int nTn = N >> 5;
  const int bx = (int)(blockIdx.x % nTn);
  const int by = (int)(blockIdx.x / nTn);
  const int tx = threadIdx.x & 31;
  const int ty = threadIdx.x >> 5;
#pragma unroll
  for (int i = 0; i < 4; ++i) {
    int r = ty + 8 * i;
    tile[r][tx] = in[(size_t)(by * 32 + r) * N + bx * 32 + tx];
  }
  __syncthreads();
#pragma unroll
  for (int i = 0; i < 4; ++i) {
    int r = ty + 8 * i;
    out[(size_t)(bx * 32 + r) * K + by * 32 + tx] = (__bf16)tile[tx][r];
  }
}

// ---------------- LayerNorm (row per block) -> bf16 ----------------
__global__ __launch_bounds__(256) void ln_bf16_kernel(
    const float* __restrict__ x, const float* __restrict__ sc,
    const float* __restrict__ sh, __bf16* __restrict__ out) {
  __shared__ float red[8];
  const int t = threadIdx.x, lane = t & 31, w = t >> 5;
  const float* xr = x + (size_t)blockIdx.x * T_D;
  float vals[8];
  float s = 0.f;
#pragma unroll
  for (int i = 0; i < 8; ++i) { vals[i] = xr[t + 256 * i]; s += vals[i]; }
#pragma unroll
  for (int o = 16; o > 0; o >>= 1) s += __shfl_xor(s, o, 32);
  if (lane == 0) red[w] = s;
  __syncthreads();
  float mean = 0.f;
#pragma unroll
  for (int i = 0; i < 8; ++i) mean += red[i];
  mean *= (1.0f / T_D);
  __syncthreads();
  float vs = 0.f;
#pragma unroll
  for (int i = 0; i < 8; ++i) { float d = vals[i] - mean; vs += d * d; }
#pragma unroll
  for (int o = 16; o > 0; o >>= 1) vs += __shfl_xor(vs, o, 32);
  if (lane == 0) red[w] = vs;
  __syncthreads();
  float var = 0.f;
#pragma unroll
  for (int i = 0; i < 8; ++i) var += red[i];
  var *= (1.0f / T_D);
  float rstd = rsqrtf(var + 1e-5f);
  __bf16* orow = out + (size_t)blockIdx.x * T_D;
#pragma unroll
  for (int i = 0; i < 8; ++i) {
    int c = t + 256 * i;
    orow[c] = (__bf16)(sc[c] * (vals[i] - mean) * rstd + sh[c]);
  }
}

// ------- WMMA GEMM: C = A(bf16,[M][K]) x Bt(bf16,[N][K])^T + bias, fused -----
// MODE 0: bf16 out.  MODE 1: f32 out = resid + C.  MODE 2: bf16 out = gelu(C).
// Double-buffered LDS, async global->LDS copies overlapped with WMMA.
template <int MODE>
__global__ __launch_bounds__(256) void gemm_wmma(
    const __bf16* __restrict__ A, const __bf16* __restrict__ Bt,
    const float* __restrict__ bias, const float* __restrict__ resid,
    float* __restrict__ outF, __bf16* __restrict__ outB, int Ndim, int Kdim) {
  __shared__ __align__(16) __bf16 As[2][128 * 40];  // [m][k], 80B rows
  __shared__ __align__(16) __bf16 Bs[2][128 * 40];  // [n][k], 80B rows
  const int t = threadIdx.x, lane = t & 31, w = t >> 5;
  const int wm = (w >> 2) << 6;
  const int wn = (w & 3) << 5;
  const int nTn = Ndim >> 7;
  const int bm = (int)(blockIdx.x / nTn) << 7;
  const int bn = (int)(blockIdx.x % nTn) << 7;

  const unsigned asB[2] = { (unsigned)(size_t)&As[0][0], (unsigned)(size_t)&As[1][0] };
  const unsigned bsB[2] = { (unsigned)(size_t)&Bs[0][0], (unsigned)(size_t)&Bs[1][0] };

  v8f acc[4][2];
#pragma unroll
  for (int i = 0; i < 4; ++i)
#pragma unroll
    for (int j = 0; j < 2; ++j)
#pragma unroll
      for (int r = 0; r < 8; ++r) acc[i][j][r] = 0.f;

  const int nk = Kdim >> 5;
  // prologue: async-copy tile 0 (A: 128x64B, B: 128x64B; 2 chunks/thread each)
#pragma unroll
  for (int j = 0; j < 2; ++j) {
    int cc = t + 256 * j;
    int row = cc >> 2, c = (cc & 3) << 4;
    async_b128(asB[0] + row * 80 + c,
               (const char*)(A + (size_t)(bm + row) * Kdim) + c);
    async_b128(bsB[0] + row * 80 + c,
               (const char*)(Bt + (size_t)(bn + row) * Kdim) + c);
  }

  for (int ki = 0; ki < nk; ++ki) {
    wait_async0();
    __syncthreads();  // tile ki visible to all; previous compute on buf^1 done
    if (ki + 1 < nk) {
      const int k0 = (ki + 1) << 5;
      const unsigned ab = asB[(ki + 1) & 1];
      const unsigned bb = bsB[(ki + 1) & 1];
#pragma unroll
      for (int j = 0; j < 2; ++j) {
        int cc = t + 256 * j;
        int row = cc >> 2, c = (cc & 3) << 4;
        async_b128(ab + row * 80 + c,
                   (const char*)(A + (size_t)(bm + row) * Kdim + k0) + c);
        async_b128(bb + row * 80 + c,
                   (const char*)(Bt + (size_t)(bn + row) * Kdim + k0) + c);
      }
    }
    const __bf16* Ab = As[ki & 1];
    const __bf16* Bb = Bs[ki & 1];
    v16bf af[4], bfr[2];
#pragma unroll
    for (int mf = 0; mf < 4; ++mf) af[mf] = fragA_ld(Ab, 40, wm + mf * 16, 0, lane);
#pragma unroll
    for (int nf = 0; nf < 2; ++nf) bfr[nf] = fragB_ld(Bb, 40, wn + nf * 16, 0, lane);
#pragma unroll
    for (int mf = 0; mf < 4; ++mf)
#pragma unroll
      for (int nf = 0; nf < 2; ++nf)
        acc[mf][nf] = wmma_bf16(af[mf], bfr[nf], acc[mf][nf]);
  }

  const int hs = (lane >> 4) << 3;
  const int nl = lane & 15;
#pragma unroll
  for (int mf = 0; mf < 4; ++mf) {
#pragma unroll
    for (int nf = 0; nf < 2; ++nf) {
      int col = bn + wn + nf * 16 + nl;
      float bv = bias[col];
#pragma unroll
      for (int r = 0; r < 8; ++r) {
        int row = bm + wm + mf * 16 + hs + r;
        size_t idx = (size_t)row * Ndim + col;
        float vv = acc[mf][nf][r] + bv;
        if (MODE == 0) {
          outB[idx] = (__bf16)vv;
        } else if (MODE == 1) {
          outF[idx] = resid[idx] + vv;
        } else {
          float g = 0.5f * vv * (1.f + erff(vv * 0.70710678118654752f));
          outB[idx] = (__bf16)g;
        }
      }
    }
  }
}

// ---------------- Flash attention (causal), bf16 in/out, WMMA -----------------
__global__ __launch_bounds__(256) void attn_wmma(
    const __bf16* __restrict__ q, const __bf16* __restrict__ k,
    const __bf16* __restrict__ v, __bf16* __restrict__ ctx) {
  extern __shared__ __align__(16) __bf16 smem[];
  __bf16* Ks  = smem;                  // [key][hd]  128x136
  __bf16* Vts = smem + 128 * 136;      // [hd][key]  128x136
  __bf16* Ps  = smem + 2 * 128 * 136;  // [qrow][key] 128x136 (per-wave rows)
  const int t = threadIdx.x, lane = t & 31, w = t >> 5;
  const int blk = (int)blockIdx.x;
  const int qt = blk & 15;
  const int h  = (blk >> 4) & 15;
  const int b  = blk >> 8;
  const int hs = (lane >> 4) << 3;
  const int nl = lane & 15;

  const size_t seqBase = (size_t)b * T_S;
  const __bf16* qB = q + (seqBase + (size_t)qt * 128) * T_D + h * T_HD;
  const __bf16* kB = k + seqBase * T_D + h * T_HD;
  const __bf16* vB = v + seqBase * T_D + h * T_HD;
  const unsigned ksBase = (unsigned)(size_t)Ks;

  v16bf qf[4];
#pragma unroll
  for (int c = 0; c < 4; ++c) qf[c] = fragA_ld(qB, T_D, w * 16, c * 32, lane);

  v8f cacc[8];
  float mprev[8], lsum[8];
#pragma unroll
  for (int nf = 0; nf < 8; ++nf)
#pragma unroll
    for (int r = 0; r < 8; ++r) cacc[nf][r] = 0.f;
#pragma unroll
  for (int r = 0; r < 8; ++r) { mprev[r] = -3.0e38f; lsum[r] = 0.f; }

  for (int kb = 0; kb <= qt; ++kb) {
    const __bf16* kSrc = kB + (size_t)(kb * 128) * T_D;
    const __bf16* vSrc = vB + (size_t)(kb * 128) * T_D;
    // K tile via async copy: 128 rows x 256B -> 8 x 16B chunks per thread
#pragma unroll
    for (int j = 0; j < 8; ++j) {
      int cc = t + 256 * j;
      int row = cc >> 4, c = (cc & 15) << 4;
      async_b128(ksBase + row * 272 + c, (const char*)(kSrc + (size_t)row * T_D) + c);
    }
    // V tile: load dwords, store transposed [hd][key]
#pragma unroll
    for (int j = 0; j < 32; ++j) {
      int f = t + 256 * j;
      int row = f >> 6, c = f & 63;
      UPK pk;
      pk.u = *(const unsigned int*)(vSrc + (size_t)row * T_D + 2 * c);
      Vts[(2 * c) * 136 + row]     = pk.h[0];
      Vts[(2 * c + 1) * 136 + row] = pk.h[1];
    }
    wait_async0();
    __syncthreads();

    // S = Q x K^T  (per wave: 16 query rows x 128 keys)
    v8f sf[8];
#pragma unroll
    for (int nf = 0; nf < 8; ++nf) {
#pragma unroll
      for (int r = 0; r < 8; ++r) sf[nf][r] = 0.f;
#pragma unroll
      for (int c = 0; c < 4; ++c)
        sf[nf] = wmma_bf16(qf[c], fragB_ld(Ks, 136, nf * 16, c * 32, lane), sf[nf]);
    }

    const float scl = 0.08838834764831845f;  // 1/sqrt(128)
    if (kb == qt) {
#pragma unroll
      for (int nf = 0; nf < 8; ++nf) {
        int kg = kb * 128 + nf * 16 + nl;
#pragma unroll
        for (int r = 0; r < 8; ++r) {
          int qg = qt * 128 + w * 16 + hs + r;
          float sv = sf[nf][r] * scl;
          sf[nf][r] = (kg > qg) ? -3.0e38f : sv;
        }
      }
    } else {
#pragma unroll
      for (int nf = 0; nf < 8; ++nf)
#pragma unroll
        for (int r = 0; r < 8; ++r) sf[nf][r] *= scl;
    }

    // online softmax, per-row; rows live in 16-lane halves
#pragma unroll
    for (int r = 0; r < 8; ++r) {
      float mx = sf[0][r];
#pragma unroll
      for (int nf = 1; nf < 8; ++nf) mx = fmaxf(mx, sf[nf][r]);
#pragma unroll
      for (int o = 8; o > 0; o >>= 1) mx = fmaxf(mx, __shfl_xor(mx, o, 16));
      float mnew  = fmaxf(mprev[r], mx);
      float alpha = __expf(mprev[r] - mnew);
      float rs = 0.f;
#pragma unroll
      for (int nf = 0; nf < 8; ++nf) {
        float e = __expf(sf[nf][r] - mnew);
        sf[nf][r] = e;
        rs += e;
      }
#pragma unroll
      for (int o = 8; o > 0; o >>= 1) rs += __shfl_xor(rs, o, 16);
      lsum[r]  = lsum[r] * alpha + rs;
      mprev[r] = mnew;
#pragma unroll
      for (int nf = 0; nf < 8; ++nf) cacc[nf][r] *= alpha;
    }

    // P (C layout) -> LDS -> A-fragment layout (wave-private region)
#pragma unroll
    for (int nf = 0; nf < 8; ++nf)
#pragma unroll
      for (int r = 0; r < 8; ++r)
        Ps[(w * 16 + hs + r) * 136 + nf * 16 + nl] = (__bf16)sf[nf][r];

    v16bf pf[4];
#pragma unroll
    for (int c = 0; c < 4; ++c) pf[c] = fragA_ld(Ps, 136, w * 16, c * 32, lane);
#pragma unroll
    for (int nf = 0; nf < 8; ++nf)
#pragma unroll
      for (int c = 0; c < 4; ++c)
        cacc[nf] = wmma_bf16(pf[c], fragB_ld(Vts, 136, nf * 16, c * 32, lane),
                             cacc[nf]);
    __syncthreads();
  }

#pragma unroll
  for (int nf = 0; nf < 8; ++nf) {
    int col = h * T_HD + nf * 16 + nl;
#pragma unroll
    for (int r = 0; r < 8; ++r) {
      int srow = qt * 128 + w * 16 + hs + r;
      float ov = cacc[nf][r] / lsum[r];
      ctx[(seqBase + srow) * T_D + col] = (__bf16)ov;
    }
  }
}

extern "C" void kernel_launch(void* const* d_in, const int* in_sizes, int n_in,
                              void* d_out, int out_size, void* d_ws,
                              size_t ws_size, hipStream_t stream) {
  (void)in_sizes; (void)n_in; (void)out_size; (void)ws_size;
  const float* x   = (const float*)d_in[0];
  const float* l1s = (const float*)d_in[1];
  const float* l1b = (const float*)d_in[2];
  const float* wq  = (const float*)d_in[3];
  const float* bq  = (const float*)d_in[4];
  const float* wk  = (const float*)d_in[5];
  const float* bk  = (const float*)d_in[6];
  const float* wv  = (const float*)d_in[7];
  const float* bv  = (const float*)d_in[8];
  const float* wo  = (const float*)d_in[9];
  const float* bo  = (const float*)d_in[10];
  const float* l2s = (const float*)d_in[11];
  const float* l2b = (const float*)d_in[12];
  const float* w1  = (const float*)d_in[13];
  const float* b1  = (const float*)d_in[14];
  const float* w2  = (const float*)d_in[15];
  const float* b2  = (const float*)d_in[16];
  float* out = (float*)d_out;

  char* ws = (char*)d_ws;
  size_t off = 0;
  auto take = [&](size_t bytes) {
    char* p = ws + off;
    off += (bytes + 255) & ~(size_t)255;
    return p;
  };
  __bf16* wqt  = (__bf16*)take((size_t)T_D * T_D * 2);   // [N][K]
  __bf16* wkt  = (__bf16*)take((size_t)T_D * T_D * 2);
  __bf16* wvt  = (__bf16*)take((size_t)T_D * T_D * 2);
  __bf16* wot  = (__bf16*)take((size_t)T_D * T_D * 2);
  __bf16* w1t  = (__bf16*)take((size_t)T_D * T_FF * 2);  // [FF][D]
  __bf16* w2t  = (__bf16*)take((size_t)T_FF * T_D * 2);  // [D][FF]
  __bf16* hb   = (__bf16*)take((size_t)T_M * T_D * 2);
  __bf16* qbuf = (__bf16*)take((size_t)T_M * T_D * 2);
  __bf16* kbuf = (__bf16*)take((size_t)T_M * T_D * 2);
  __bf16* vbuf = (__bf16*)take((size_t)T_M * T_D * 2);
  __bf16* ctxb = (__bf16*)take((size_t)T_M * T_D * 2);
  float*  xmid = (float*)take((size_t)T_M * T_D * 4);
  __bf16* h2b  = (__bf16*)take((size_t)T_M * T_D * 2);
  __bf16* a1b  = (__bf16*)take((size_t)T_M * T_FF * 2);

  // 1) weight convert + transpose: fp32 [K][N] -> bf16 [N][K]
  transpose_to_bf16<<<(T_D / 32) * (T_D / 32), 256, 0, stream>>>(wq, wqt, T_D, T_D);
  transpose_to_bf16<<<(T_D / 32) * (T_D / 32), 256, 0, stream>>>(wk, wkt, T_D, T_D);
  transpose_to_bf16<<<(T_D / 32) * (T_D / 32), 256, 0, stream>>>(wv, wvt, T_D, T_D);
  transpose_to_bf16<<<(T_D / 32) * (T_D / 32), 256, 0, stream>>>(wo, wot, T_D, T_D);
  transpose_to_bf16<<<(T_D / 32) * (T_FF / 32), 256, 0, stream>>>(w1, w1t, T_D, T_FF);
  transpose_to_bf16<<<(T_FF / 32) * (T_D / 32), 256, 0, stream>>>(w2, w2t, T_FF, T_D);

  // 2) LN1
  ln_bf16_kernel<<<T_M, 256, 0, stream>>>(x, l1s, l1b, hb);

  // 3) QKV projections (M=4096, N=2048, K=2048)
  gemm_wmma<0><<<512, 256, 0, stream>>>(hb, wqt, bq, nullptr, nullptr, qbuf, T_D, T_D);
  gemm_wmma<0><<<512, 256, 0, stream>>>(hb, wkt, bk, nullptr, nullptr, kbuf, T_D, T_D);
  gemm_wmma<0><<<512, 256, 0, stream>>>(hb, wvt, bv, nullptr, nullptr, vbuf, T_D, T_D);

  // 4) causal flash attention: B*H*(S/128) = 512 blocks, 102KB dynamic LDS
  attn_wmma<<<512, 256, 3 * 128 * 136 * (int)sizeof(__bf16), stream>>>(qbuf, kbuf,
                                                                       vbuf, ctxb);

  // 5) O projection + residual (fp32 out)
  gemm_wmma<1><<<512, 256, 0, stream>>>(ctxb, wot, bo, x, xmid, nullptr, T_D, T_D);

  // 6) LN2
  ln_bf16_kernel<<<T_M, 256, 0, stream>>>(xmid, l2s, l2b, h2b);

  // 7) FFN1 + exact GELU (M=4096, N=8192, K=2048)
  gemm_wmma<2><<<2048, 256, 0, stream>>>(h2b, w1t, b1, nullptr, nullptr, a1b, T_FF, T_D);

  // 8) FFN2 + residual (M=4096, N=2048, K=8192) -> d_out fp32
  gemm_wmma<1><<<512, 256, 0, stream>>>(a1b, w2t, b2, xmid, out, nullptr, T_D, T_FF);
}
